// FiGATv2_Layer_33998961115705
// MI455X (gfx1250) — matrix operand order repaired and checked by reading, not compiled, not run on hardware
//
#include <hip/hip_runtime.h>
#include <hip/hip_bf16.h>

typedef __attribute__((ext_vector_type(16))) _Float16 v16h;
typedef __attribute__((ext_vector_type(8)))  float    v8f;
typedef __attribute__((ext_vector_type(4)))  unsigned u32x4;
typedef __attribute__((ext_vector_type(8)))  int      i32x8;
typedef __attribute__((ext_vector_type(4)))  int      i32x4;

#define NN   24      // N
#define NP   32      // N padded to 2x16 tiles
#define DD   64      // D
#define D2   32      // D/2
#define ND2  768     // N*D2
#define GR   192     // 3*D
#define LL   3
#define BB   1024
#define WELEM (2 * DD * D2)   // 4096 f32 per layer-batch W

// LDS strides (16B-aligned contiguous runs for ds b128; not multiples of 64 banks)
#define H16S 80      // stride for sh16 / sAo16 (f16, row-major M x K)
#define ATTS 48      // stride for sAtt16
#define UVS  36      // stride for sU / sV (f32)
#define ES   33      // stride for sE (f32)
#define GS   196     // stride for sGi / sGh (f32)

// ---- WMMA operand loaders (contiguous k order: element e of lane half h holds k = 16h + e) ----
static __device__ __forceinline__ v16h wmma_load_A(const _Float16* buf, int stride,
                                                   int m0, int k0) {
  const int lane = threadIdx.x & 31;
  const int m  = m0 + (lane & 15);
  const int hh = (lane >> 4) << 4;           // 0 or 16
  const _Float16* p = buf + m * stride + k0 + hh;
  v16h a;
#pragma unroll
  for (int e = 0; e < 16; ++e) a[e] = p[e];
  return a;
}

// B 32x16 f16 from a TRANSPOSED (n-major) buffer; columns >= nmax read as 0.
static __device__ __forceinline__ v16h wmma_load_BT(const _Float16* bufT, int stride,
                                                    int k0, int n0, int nmax) {
  const int lane = threadIdx.x & 31;
  const int n  = n0 + (lane & 15);
  const int kh = k0 + ((lane >> 4) << 4);
  v16h b;
  if (n < nmax) {
    const _Float16* p = bufT + n * stride + kh;
#pragma unroll
    for (int e = 0; e < 16; ++e) b[e] = p[e];
  } else {
#pragma unroll
    for (int e = 0; e < 16; ++e) b[e] = (_Float16)0.0f;
  }
  return b;
}

// B operand gathered from global f32 (n-major, w[n*K + k]); L2-resident GRU weights.
static __device__ __forceinline__ v16h wmma_load_BT_gf32(const float* __restrict__ bufT,
                                                         int stride, int k0, int n0) {
  const int lane = threadIdx.x & 31;
  const int n  = n0 + (lane & 15);
  const int kh = k0 + ((lane >> 4) << 4);
  const float* p = bufT + n * stride + kh;
  v16h b;
#pragma unroll
  for (int e = 0; e < 16; ++e) b[e] = (_Float16)p[e];
  return b;
}

// C/D 16x16 f32: vgpr e -> row m0 + (lane>>4)*8 + e, col n0 + (lane&15)
static __device__ __forceinline__ void wmma_store_C(float* buf, int stride,
                                                    int m0, int n0, v8f c, int mmax) {
  const int lane = threadIdx.x & 31;
  const int n  = n0 + (lane & 15);
  const int mb = m0 + ((lane >> 4) << 3);
#pragma unroll
  for (int e = 0; e < 8; ++e) {
    const int m = mb + e;
    if (m < mmax) buf[m * stride + n] = c[e];
  }
}

#define WMMA(a, b, c) \
  __builtin_amdgcn_wmma_f32_16x16x32_f16(false, (a), false, (b), (short)0, (c), false, false)

// ---- TDM: async DMA of one (128 x 32) f32 W tile, global -> LDS ----
// Descriptor per CDNA5 ISA:
//  g0: [1:0]=count=1, [63:32]=lds_addr, [120:64]=global_addr, [127:126]=type=2
//  g1: data_size=2(4B); tensor_dim0=32, tensor_dim1=128, tile_dim0=32,
//      tile_dim1=128, tensor_dim0_stride=32 (data_size units)
// Toolchain uses the 6-arg builtin: (g0, g1, g2:i32x4, g3:i32x4, gx:i32x8, cpol)
static __device__ __forceinline__ void tdm_load_w(unsigned lds_addr, const float* gsrc) {
  const unsigned long long ga = (unsigned long long)(uintptr_t)gsrc;
  u32x4 g0 = { 1u,
               lds_addr,
               (unsigned)ga,
               ((unsigned)((ga >> 32) & 0x01FFFFFFu)) | 0x80000000u };
  i32x8 g1 = { (int)0x00020000,        // workgroup_mask=0, data_size=4B
               (int)(32u << 16),       // tensor_dim0 = 32 (in [31:16])
               (int)(128u << 16),      // tensor_dim0[31:16]=0, tensor_dim1 = 128
               (int)(32u << 16),       // tensor_dim1[31:16]=0, tile_dim0 = 32
               (int)128,               // tile_dim1 = 128, tile_dim2 = 0
               (int)32,                // tensor_dim0_stride = 32
               0, 0 };
  i32x4 z4 = { 0, 0, 0, 0 };
  i32x8 z8 = { 0, 0, 0, 0, 0, 0, 0, 0 };
  __builtin_amdgcn_tensor_load_to_lds(g0, g1, z4, z4, z8, 0);
}

__global__ __launch_bounds__(128) void figat_kernel(
    const float* __restrict__ x,    // feature_emb (B,N,D)
    const float* __restrict__ Ws,   // (L,B,2D,D2)
    const float* __restrict__ As,   // (L,N*D2,N)
    const float* __restrict__ wih,  // (3D,D)
    const float* __restrict__ whh,  // (3D,D)
    const float* __restrict__ bih,  // (3D)
    const float* __restrict__ bhh,  // (3D)
    float* __restrict__ out)        // (B,N,D)
{
  extern __shared__ char smem[];
  float* sWst = (float*)smem;                    // 2 * WELEM  TDM double buffer (16KB x2)
  float* sh   = sWst + 2 * WELEM;                // NP*DD   h (f32), pad rows = 0
  float* sx   = sh + NP * DD;                    // NP*DD   residual copy
  float* sU   = sx + NP * DD;                    // NP*UVS
  float* sV   = sU + NP * UVS;                   // NP*UVS
  float* sE   = sV + NP * UVS;                   // NP*ES   raw attention logits
  float* sGi  = sE + NP * ES;                    // NN*GS   GRU input gates
  float* sGh  = sGi + NN * GS;                   // NN*GS   GRU hidden gates
  _Float16* sh16   = (_Float16*)(sGh + NN * GS); // NP*H16S  h row-major f16 (A operand)
  _Float16* sh16T  = sh16 + NP * H16S;           // DD*NP    h transposed f16 (B operand)
  _Float16* sW16T  = sh16T + DD * NP;            // D2*(2*DD) W transposed f16
  _Float16* sAsT   = sW16T + D2 * 2 * DD;        // NN*ND2   As transposed f16
  _Float16* sAtt16 = sAsT + NN * ND2;            // NP*ATTS  att f16 (A operand)
  _Float16* sAo16  = sAtt16 + NP * ATTS;         // NP*H16S  a_out f16 (A operand)

  const int b    = blockIdx.x;
  const int tid  = threadIdx.x;
  const int wave = __builtin_amdgcn_readfirstlane(tid >> 5);  // SGPR-uniform wave id
  const int lane = tid & 31;

  // ---- kick off TDM for layer 0's per-batch W while we initialize h ----
  if (wave == 0)
    tdm_load_w((unsigned)(uintptr_t)sWst,
               Ws + ((size_t)0 * BB + b) * WELEM);

  for (int i = tid; i < NP * DD; i += 128) {
    const int r = i / DD, c = i - r * DD;
    const float v = (r < NN) ? x[((size_t)b * NN + r) * DD + c] : 0.0f;
    sh[i] = v;
    sx[i] = v;
    sh16[r * H16S + c] = (_Float16)v;
    sh16T[c * NP + r]  = (_Float16)v;
  }
  __syncthreads();

  for (int l = 0; l < LL; ++l) {
    // ---- consume W[l] DMA; prefetch W[l+1] into the other buffer ----
    if (wave == 0) __builtin_amdgcn_s_wait_tensorcnt(0);
    __syncthreads();                       // W[l] visible to all; old readers of buf[(l+1)&1] done
    if (wave == 0 && l + 1 < LL)
      tdm_load_w((unsigned)(uintptr_t)(sWst + ((l + 1) & 1) * WELEM),
                 Ws + ((size_t)(l + 1) * BB + b) * WELEM);

    // L2 prefetch of next layer's As while this layer computes
    if (l + 1 < LL) {
      const char* nx = (const char*)(As + (size_t)(l + 1) * ND2 * NN);
      for (int off = tid * 128; off < (int)(ND2 * NN * sizeof(float)); off += 128 * 128)
        __builtin_prefetch(nx + off, 0, 1);
    }

    // ---- stage As[l] transposed (n-major) as f16; convert W (LDS f32 -> f16^T) ----
    const float* Asl = As + (size_t)l * ND2 * NN;
    for (int i = tid; i < ND2 * NN; i += 128) {
      const int m = i / NN, n = i - m * NN;
      sAsT[n * ND2 + m] = (_Float16)Asl[i];
    }
    const float* Wl = sWst + (l & 1) * WELEM;
    for (int i = tid; i < WELEM; i += 128) {
      const int k = i / D2, n = i - k * D2;
      sW16T[n * (2 * DD) + k] = (_Float16)Wl[i];
    }
    __syncthreads();

    // ---- U = h @ W_top, V = h @ W_bot   (pairwise-concat GEMM, factorized) ----
    for (int t = wave; t < 8; t += 4) {
      const int isV = t >> 2;
      const int tt  = t & 3;
      const int m0  = (tt >> 1) << 4;
      const int n0  = (tt & 1) << 4;
      v8f acc = {};
#pragma unroll
      for (int kk = 0; kk < DD; kk += 32) {
        v16h a  = wmma_load_A(sh16, H16S, m0, kk);
        v16h bt = wmma_load_BT(sW16T, 2 * DD, kk + (isV ? DD : 0), n0, D2);
        acc = WMMA(a, bt, acc);
      }
      wmma_store_C(isV ? sV : sU, UVS, m0, n0, acc, NP);
    }
    __syncthreads();

    // ---- e = leaky_relu(U_i + V_j) @ As   (24x768 @ 768x24), A built on the fly ----
    {
      const int m0 = (wave >> 1) << 4;
      const int n0 = (wave & 1) << 4;
      const int m  = m0 + (lane & 15);
      const int hh = (lane >> 4) << 4;
      v8f acc = {};
      for (int j = 0; j < NN; ++j) {
        v16h a;
#pragma unroll
        for (int e = 0; e < 16; ++e) {
          const int c  = e + hh;
          const float v = sU[m * UVS + c] + sV[j * UVS + c];
          a[e] = (_Float16)(v > 0.0f ? v : 0.01f * v);     // leaky_relu(0.01)
        }
        v16h bt = wmma_load_BT(sAsT, ND2, j * D2, n0, NN);
        acc = WMMA(a, bt, acc);
      }
      wmma_store_C(sE, ES, m0, n0, acc, NP);
    }
    for (int i = tid; i < NP * ATTS; i += 128) sAtt16[i] = (_Float16)0.0f;
    __syncthreads();

    // ---- mask diag, softmax over axis=1 (normalize each column over rows) ----
    if (tid < NN) {
      const int j = tid;
      float vals[NN];
      float mx = -1e30f;
#pragma unroll
      for (int i = 0; i < NN; ++i) {
        const float v = (i == j) ? -1e12f : sE[i * ES + j];
        vals[i] = v;
        mx = fmaxf(mx, v);
      }
      float s = 0.0f;
#pragma unroll
      for (int i = 0; i < NN; ++i) { vals[i] = __expf(vals[i] - mx); s += vals[i]; }
      const float inv = 1.0f / s;
#pragma unroll
      for (int i = 0; i < NN; ++i) sAtt16[i * ATTS + j] = (_Float16)(vals[i] * inv);
    }
    __syncthreads();

    // ---- a_out = elu(att @ h)   (24x24 @ 24x64, K padded to 32 with zeros) ----
    for (int t = wave; t < 8; t += 4) {
      const int m0 = (t & 1) << 4;
      const int n0 = (t >> 1) << 4;
      v16h a  = wmma_load_A(sAtt16, ATTS, m0, 0);
      v16h bt = wmma_load_BT(sh16T, NP, 0, n0, DD);
      v8f acc = {};
      acc = WMMA(a, bt, acc);
      const int n  = n0 + (lane & 15);
      const int mb = m0 + ((lane >> 4) << 3);
#pragma unroll
      for (int e = 0; e < 8; ++e) {
        float v = acc[e];
        v = (v > 0.0f) ? v : (__expf(v) - 1.0f);           // elu
        sAo16[(mb + e) * H16S + n] = (_Float16)v;
      }
    }
    __syncthreads();

    // ---- GRU gates: gi = a_out @ wih^T, gh = h @ whh^T   (24x64 @ 64x192) ----
    for (int t = wave; t < 48; t += 4) {
      const int g  = t / 24;
      const int tt = t - g * 24;
      const int m0 = (tt & 1) << 4;
      const int n0 = (tt >> 1) << 4;
      const _Float16* Abuf = g ? sh16 : sAo16;
      const float* Wg = g ? whh : wih;
      v8f acc = {};
#pragma unroll
      for (int kk = 0; kk < DD; kk += 32) {
        v16h a  = wmma_load_A(Abuf, H16S, m0, kk);
        v16h bt = wmma_load_BT_gf32(Wg, DD, kk, n0);
        acc = WMMA(a, bt, acc);
      }
      wmma_store_C(g ? sGh : sGi, GS, m0, n0, acc, NN);
    }
    __syncthreads();

    // ---- GRU elementwise + residual; refresh h in all three layouts ----
    for (int i = tid; i < NN * DD; i += 128) {
      const int r = i / DD, d = i - r * DD;
      const float ir  = sGi[r * GS + d]          + bih[d];
      const float iz  = sGi[r * GS + DD + d]     + bih[DD + d];
      const float inn = sGi[r * GS + 2 * DD + d] + bih[2 * DD + d];
      const float hr  = sGh[r * GS + d]          + bhh[d];
      const float hz  = sGh[r * GS + DD + d]     + bhh[DD + d];
      const float hn  = sGh[r * GS + 2 * DD + d] + bhh[2 * DD + d];
      const float rg = 1.0f / (1.0f + __expf(-(ir + hr)));
      const float zg = 1.0f / (1.0f + __expf(-(iz + hz)));
      const float ng = tanhf(inn + rg * hn);
      const float hnew = (1.0f - zg) * ng + zg * sh[i] + sx[i];
      sh[i] = hnew;
      sh16[r * H16S + d] = (_Float16)hnew;
      sh16T[d * NP + r]  = (_Float16)hnew;
    }
    __syncthreads();
  }

  for (int i = tid; i < NN * DD; i += 128)
    out[(size_t)b * NN * DD + i] = sh[i];
}

extern "C" void kernel_launch(void* const* d_in, const int* in_sizes, int n_in,
                              void* d_out, int out_size, void* d_ws, size_t ws_size,
                              hipStream_t stream) {
  (void)in_sizes; (void)n_in; (void)out_size; (void)d_ws; (void)ws_size;
  const float* x   = (const float*)d_in[0];
  // d_in[1] (w_attn) is dead in the reference: g = softmax(alpha) is never used.
  const float* Ws  = (const float*)d_in[2];
  const float* As  = (const float*)d_in[3];
  const float* wih = (const float*)d_in[4];
  const float* whh = (const float*)d_in[5];
  const float* bih = (const float*)d_in[6];
  const float* bhh = (const float*)d_in[7];
  float* out = (float*)d_out;

  const size_t fElems = (size_t)2 * WELEM + (size_t)NP * DD * 2 + (size_t)NP * UVS * 2 +
                        (size_t)NP * ES + (size_t)NN * GS * 2;
  const size_t hElems = (size_t)NP * H16S * 2 + (size_t)DD * NP +
                        (size_t)D2 * 2 * DD + (size_t)NN * ND2 + (size_t)NP * ATTS;
  const size_t shmem = fElems * 4 + hElems * 2;   // ~159 KB -> still 2 WGs per 320 KB WGP

  figat_kernel<<<dim3(BB), dim3(128), shmem, stream>>>(x, Ws, As, wih, whh, bih, bhh, out);
}
